// BERT_JOINT_87531433492489
// MI455X (gfx1250) — compile-verified, matrix operand order checked by reading, not run on hardware
//
#include <hip/hip_runtime.h>
#include <math.h>

typedef __attribute__((ext_vector_type(16))) _Float16 v16h;
typedef __attribute__((ext_vector_type(8)))  _Float16 v8h;
typedef __attribute__((ext_vector_type(8)))  float    v8f;

#define B_ 4
#define S_ 256
#define D_ 768
#define T_ 9
#define L_ 30
#define R_ 12
#define H_ 798
#define HP 800   // H padded to multiple of 32 (25 k-chunks of 32)

// Native CDNA5 TRANS32 tanh (co-executes with XDL WMMA); guarded fallback.
#if __has_builtin(__builtin_amdgcn_tanhf)
__device__ __forceinline__ float fast_tanh(float x) { return __builtin_amdgcn_tanhf(x); }
#elif __has_builtin(__builtin_amdgcn_tanh_f32)
__device__ __forceinline__ float fast_tanh(float x) { return __builtin_amdgcn_tanh_f32(x); }
#else
__device__ __forceinline__ float fast_tanh(float x) { return tanhf(x); }
#endif

// ---------------- kernel 0: build padded, transposed f16 weight copies ----------------
// UT/WT: [HP][HP] halfs with UT[n][k] = U_w[k][n]  (zeros in pad region)
// VT:    [16][HP] halfs with VT[r][k] = V_w[k][r]  (r<12, else 0)
__global__ void prep_kernel(const float* __restrict__ Uw, const float* __restrict__ Ww,
                            const float* __restrict__ Vw,
                            _Float16* __restrict__ UT, _Float16* __restrict__ WT,
                            _Float16* __restrict__ VT) {
  int t = blockIdx.x * blockDim.x + threadIdx.x;
  if (t < HP * HP) {
    int n = t / HP, k = t % HP;
    bool ok = (n < H_) && (k < H_);
    UT[t] = (_Float16)(ok ? Uw[k * H_ + n] : 0.f);
    WT[t] = (_Float16)(ok ? Ww[k * H_ + n] : 0.f);
  }
  if (t < 16 * HP) {
    int r = t / HP, k = t % HP;
    VT[t] = (_Float16)((r < R_ && k < H_) ? Vw[k * R_ + r] : 0.f);
  }
}

// ---------------- kernel 1: ner = embed @ linear_w + linear_b  [B,S,T] ----------------
__global__ void ner_kernel(const float* __restrict__ embed, const float* __restrict__ lw,
                           const float* __restrict__ lb, float* __restrict__ ner) {
  int bs   = blockIdx.x;
  int wave = threadIdx.x >> 5;   // tag 0..8
  int lane = threadIdx.x & 31;
  const float* e = embed + (size_t)bs * D_;
  float sum = 0.f;
  for (int d = lane; d < D_; d += 32) sum += e[d] * lw[d * T_ + wave];
  for (int off = 16; off > 0; off >>= 1) sum += __shfl_down(sum, off, 32);
  if (lane == 0) ner[bs * T_ + wave] = sum + lb[wave];
}

// ---------------- kernel 2: Viterbi decode, one wave per batch ----------------
__global__ void viterbi_kernel(const float* __restrict__ ner, const float* __restrict__ st,
                               const float* __restrict__ tr, const float* __restrict__ et,
                               int* __restrict__ tags_out, int* __restrict__ tags_ws) {
  __shared__ int hist[(S_ - 1) * T_];
  int b = blockIdx.x;
  int j = threadIdx.x;                      // lanes 0..8 own tags
  const float* nb = ner + (size_t)b * S_ * T_;
  float score = (j < T_) ? st[j] + nb[j] : -1e30f;
  for (int s = 1; s < S_; ++s) {
    float best = -1e30f; int bp = 0;
    for (int p = 0; p < T_; ++p) {
      float sp = __shfl(score, p, 32);
      float v  = sp + ((j < T_) ? tr[p * T_ + j] : 0.f);
      if (v > best) { best = v; bp = p; }
    }
    if (j < T_) hist[(s - 1) * T_ + j] = bp;
    score = (j < T_) ? best + nb[s * T_ + j] : -1e30f;
  }
  float fin = (j < T_) ? score + et[j] : -1e30f;
  int last = 0; float bv = -1e30f;
  for (int p = 0; p < T_; ++p) {
    float v = __shfl(fin, p, 32);
    if (v > bv) { bv = v; last = p; }
  }
  __syncthreads();
  if (j == 0) {
    int tag = last;
    tags_out[b * S_ + S_ - 1] = tag; tags_ws[b * S_ + S_ - 1] = tag;
    for (int s = S_ - 2; s >= 0; --s) {
      tag = hist[s * T_ + tag];
      tags_out[b * S_ + s] = tag; tags_ws[b * S_ + s] = tag;
    }
  }
}

// ---------------- kernel 3: hx = concat(embed, label_emb[tags], pad) [B*S][HP] ----------------
__global__ void hx_kernel(const float* __restrict__ embed, const float* __restrict__ lemb,
                          const int* __restrict__ tags, float* __restrict__ hx) {
  int bs = blockIdx.x;
  int tag = tags[bs];
  const float* e = embed + (size_t)bs * D_;
  float* o = hx + (size_t)bs * HP;
  for (int h = threadIdx.x; h < HP; h += blockDim.x) {
    float v;
    if (h < D_)      v = e[h];
    else if (h < H_) v = lemb[tag * L_ + (h - D_)];
    else             v = 0.f;
    o[h] = v;
  }
}

// load a 16x32 f16 A-fragment from a row-major f32 row (ISA 7.12.2 lane striping)
__device__ __forceinline__ v16h load_a_frag_f32(const float* row, int k0, int k1) {
  const float4* p0 = reinterpret_cast<const float4*>(row + k0);
  const float4* p1 = reinterpret_cast<const float4*>(row + k1);
  float4 x0 = p0[0], x1 = p0[1], y0 = p1[0], y1 = p1[1];
  v16h a;
  a[0]=(_Float16)x0.x;  a[1]=(_Float16)x0.y;  a[2]=(_Float16)x0.z;  a[3]=(_Float16)x0.w;
  a[4]=(_Float16)x1.x;  a[5]=(_Float16)x1.y;  a[6]=(_Float16)x1.z;  a[7]=(_Float16)x1.w;
  a[8]=(_Float16)y0.x;  a[9]=(_Float16)y0.y;  a[10]=(_Float16)y0.z; a[11]=(_Float16)y0.w;
  a[12]=(_Float16)y1.x; a[13]=(_Float16)y1.y; a[14]=(_Float16)y1.z; a[15]=(_Float16)y1.w;
  return a;
}

// load a 32x16 f16 B-fragment from a [n][k] half row
__device__ __forceinline__ v16h load_b_frag_f16(const _Float16* row, int k0, int k1) {
  v8h b0 = *reinterpret_cast<const v8h*>(row + k0);
  v8h b1 = *reinterpret_cast<const v8h*>(row + k1);
  v16h b;
  for (int p = 0; p < 8; ++p) { b[p] = b0[p]; b[8 + p] = b1[p]; }
  return b;
}

// tanh(u+w) A-fragment from f32 global w-row and LDS u-row
__device__ __forceinline__ v16h tanh_a_frag(const float* __restrict__ wrow,
                                            const float* __restrict__ sU,
                                            int k0, int k1) {
  float4 wa = *reinterpret_cast<const float4*>(wrow + k0);
  float4 wb = *reinterpret_cast<const float4*>(wrow + k0 + 4);
  float4 wc = *reinterpret_cast<const float4*>(wrow + k1);
  float4 wd = *reinterpret_cast<const float4*>(wrow + k1 + 4);
  float4 ua = *reinterpret_cast<const float4*>(sU + k0);
  float4 ub = *reinterpret_cast<const float4*>(sU + k0 + 4);
  float4 uc = *reinterpret_cast<const float4*>(sU + k1);
  float4 ud = *reinterpret_cast<const float4*>(sU + k1 + 4);
  v16h a;
  a[0]  = (_Float16)fast_tanh(ua.x + wa.x);  a[1]  = (_Float16)fast_tanh(ua.y + wa.y);
  a[2]  = (_Float16)fast_tanh(ua.z + wa.z);  a[3]  = (_Float16)fast_tanh(ua.w + wa.w);
  a[4]  = (_Float16)fast_tanh(ub.x + wb.x);  a[5]  = (_Float16)fast_tanh(ub.y + wb.y);
  a[6]  = (_Float16)fast_tanh(ub.z + wb.z);  a[7]  = (_Float16)fast_tanh(ub.w + wb.w);
  a[8]  = (_Float16)fast_tanh(uc.x + wc.x);  a[9]  = (_Float16)fast_tanh(uc.y + wc.y);
  a[10] = (_Float16)fast_tanh(uc.z + wc.z);  a[11] = (_Float16)fast_tanh(uc.w + wc.w);
  a[12] = (_Float16)fast_tanh(ud.x + wd.x);  a[13] = (_Float16)fast_tanh(ud.y + wd.y);
  a[14] = (_Float16)fast_tanh(ud.z + wd.z);  a[15] = (_Float16)fast_tanh(ud.w + wd.w);
  return a;
}

// ---------------- kernel 4: u = hx@U + Ub, w = hx@W + Wb via WMMA ----------------
// one wave per 16x32 output tile (two 16x16 N-tiles sharing one A fragment)
__global__ void uw_gemm_kernel(const float* __restrict__ hx,
                               const _Float16* __restrict__ UT, const _Float16* __restrict__ WT,
                               const float* __restrict__ Ub, const float* __restrict__ Wb,
                               float* __restrict__ ubuf, float* __restrict__ wbuf) {
  const int MT = (B_ * S_) / 16;   // 64
  const int NP = HP / 32;          // 25 pairs of N-tiles
  int wg  = blockIdx.x * (blockDim.x >> 5) + (threadIdx.x >> 5);
  int mat = wg / (MT * NP);
  int rem = wg % (MT * NP);
  int mt = rem / NP, np = rem % NP;
  int lane = threadIdx.x & 31;
  int hi = lane >> 4;              // K-halve select
  int l  = lane & 15;
  int n0 = np * 32;

  const _Float16* BT   = mat ? WT : UT;
  const float*    bias = mat ? Wb : Ub;
  float*          outp = mat ? wbuf : ubuf;

  const float*    arow  = hx + (size_t)(mt * 16 + l) * HP;
  const _Float16* brow0 = BT + (size_t)(n0 + l) * HP;
  const _Float16* brow1 = BT + (size_t)(n0 + 16 + l) * HP;

  v8f acc0 = {}, acc1 = {};
  for (int c = 0; c < HP; c += 32) {
    int k0 = c + hi * 8;
    int k1 = c + 16 + hi * 8;
    v16h a  = load_a_frag_f32(arow, k0, k1);       // shared by both N-tiles
    v16h b0 = load_b_frag_f16(brow0, k0, k1);
    v16h b1 = load_b_frag_f16(brow1, k0, k1);
    acc0 = __builtin_amdgcn_wmma_f32_16x16x32_f16(false, a, false, b0, (short)0, acc0, false, false);
    acc1 = __builtin_amdgcn_wmma_f32_16x16x32_f16(false, a, false, b1, (short)0, acc1, false, false);
  }
  int na = n0 + l, nb = n0 + 16 + l;               // D-layout: N = lane&15
  float bva = (na < H_) ? bias[na] : 0.f;
  float bvb = (nb < H_) ? bias[nb] : 0.f;
  for (int v = 0; v < 8; ++v) {
    int m = mt * 16 + v + hi * 8;                  // D-layout: M = vgpr + 8*hi
    outp[(size_t)m * HP + na] = acc0[v] + bva;
    outp[(size_t)m * HP + nb] = acc1[v] + bvb;
  }
}

// ---------------- kernel 5: fused tanh + einsum via WMMA ----------------
// scores[b,i,j,r] = sum_h tanh(u[b,i,h]+w[b,j,h]) * V[h,r]
// one block per (b,i); each of 8 waves owns two 16-wide j-tiles sharing one V fragment.
__global__ void pair_kernel(const float* __restrict__ u, const float* __restrict__ w,
                            const _Float16* __restrict__ VT,
                            float* __restrict__ scores, const int* __restrict__ none_idx_p) {
  __shared__ __align__(16) _Float16 sVT[16 * HP];  // V^T in f16: 25.6 KB
  __shared__ __align__(16) float    sU[HP];        // u[b,i,:]:    3.2 KB
  int blk  = blockIdx.x;
  int b    = blk / S_;
  int i    = blk % S_;
  int wave = threadIdx.x >> 5;
  int lane = threadIdx.x & 31;
  int hi   = lane >> 4;
  int l    = lane & 15;
  int j0a  = wave * 32;        // first j-tile
  int j0b  = j0a + 16;         // second j-tile

  {
    const int4* src = reinterpret_cast<const int4*>(VT);
    int4*       dst = reinterpret_cast<int4*>(sVT);
    for (int t = threadIdx.x; t < (16 * HP) / 8; t += blockDim.x) dst[t] = src[t];
    const float* ur = u + (size_t)(b * S_ + i) * HP;
    for (int t = threadIdx.x; t < HP; t += blockDim.x) sU[t] = ur[t];
  }
  __syncthreads();

  const float*    wrowA = w + (size_t)(b * S_ + j0a + l) * HP;  // A row = j-row of tile
  const float*    wrowB = w + (size_t)(b * S_ + j0b + l) * HP;
  const _Float16* vrow  = sVT + l * HP;                         // B row = relation column

  __builtin_prefetch(wrowA, 0, 0);                 // global_prefetch: warm L2->L0 for w stream
  __builtin_prefetch(wrowB, 0, 0);

  v8f acc0 = {}, acc1 = {};
  for (int c = 0; c < HP; c += 32) {
    int k0 = c + hi * 8;
    int k1 = c + 16 + hi * 8;
    __builtin_prefetch(wrowA + c + 128, 0, 0);     // speculative; OOB dropped silently
    __builtin_prefetch(wrowB + c + 128, 0, 0);
    v16h a0 = tanh_a_frag(wrowA, sU, k0, k1);
    v16h a1 = tanh_a_frag(wrowB, sU, k0, k1);
    v16h bf = load_b_frag_f16(vrow, k0, k1);       // shared by both j-tiles
    acc0 = __builtin_amdgcn_wmma_f32_16x16x32_f16(false, a0, false, bf, (short)0, acc0, false, false);
    acc1 = __builtin_amdgcn_wmma_f32_16x16x32_f16(false, a1, false, bf, (short)0, acc1, false, false);
  }
  int r = l;                                       // D-layout: N = relation index
  if (r < R_) {
    int nidx = *none_idx_p;
    float* orow = scores + (size_t)(b * S_ + i) * S_ * R_;
    for (int v = 0; v < 8; ++v) {
      int m  = v + hi * 8;
      int ja = j0a + m, jb = j0b + m;
      float va = acc0[v], vb = acc1[v];
      if (ja == i && r == nidx) va = 100.0f;       // diagonal override fused into epilogue
      if (jb == i && r == nidx) vb = 100.0f;
      orow[(size_t)ja * R_ + r] = va;
      orow[(size_t)jb * R_ + r] = vb;
    }
  }
}

extern "C" void kernel_launch(void* const* d_in, const int* in_sizes, int n_in,
                              void* d_out, int out_size, void* d_ws, size_t ws_size,
                              hipStream_t stream) {
  const float* embed = (const float*)d_in[0];
  const float* lw    = (const float*)d_in[1];
  const float* lb    = (const float*)d_in[2];
  const float* st    = (const float*)d_in[3];
  const float* tr    = (const float*)d_in[4];
  const float* et    = (const float*)d_in[5];
  const float* lemb  = (const float*)d_in[6];
  const float* Uw    = (const float*)d_in[7];
  const float* Ub    = (const float*)d_in[8];
  const float* Ww    = (const float*)d_in[9];
  const float* Wb    = (const float*)d_in[10];
  const float* Vw    = (const float*)d_in[11];
  const int*   nidx  = (const int*)d_in[12];

  char* ws = (char*)d_ws;
  size_t off = 0;
  auto alloc = [&](size_t bytes) {
    void* p = ws + off;
    off = (off + bytes + 255) & ~(size_t)255;
    return p;
  };
  float*    ner     = (float*)alloc((size_t)B_ * S_ * T_ * sizeof(float));
  int*      tags_ws = (int*)  alloc((size_t)B_ * S_ * sizeof(int));
  float*    hx      = (float*)alloc((size_t)B_ * S_ * HP * sizeof(float));
  float*    ubuf    = (float*)alloc((size_t)B_ * S_ * HP * sizeof(float));
  float*    wbuf    = (float*)alloc((size_t)B_ * S_ * HP * sizeof(float));
  _Float16* UT      = (_Float16*)alloc((size_t)HP * HP * sizeof(_Float16));
  _Float16* WT      = (_Float16*)alloc((size_t)HP * HP * sizeof(_Float16));
  _Float16* VT      = (_Float16*)alloc((size_t)16 * HP * sizeof(_Float16));

  int*   tags_out = (int*)d_out;                 // first B*S elements: int32 tags
  float* scores   = (float*)d_out + B_ * S_;     // then [B,S,S,R] float scores

  prep_kernel<<<(HP * HP + 255) / 256, 256, 0, stream>>>(Uw, Ww, Vw, UT, WT, VT);
  ner_kernel<<<B_ * S_, 288, 0, stream>>>(embed, lw, lb, ner);
  viterbi_kernel<<<B_, 32, 0, stream>>>(ner, st, tr, et, tags_out, tags_ws);
  hx_kernel<<<B_ * S_, 256, 0, stream>>>(embed, lemb, tags_ws, hx);
  uw_gemm_kernel<<<(64 * 25 * 2) / 8, 256, 0, stream>>>(hx, UT, WT, Ub, Wb, ubuf, wbuf);
  pair_kernel<<<B_ * S_, 256, 0, stream>>>(ubuf, wbuf, VT, scores, nidx);
}
